// Head_32220844654678
// MI455X (gfx1250) — matrix operand order repaired and checked by reading, not compiled, not run on hardware
//
#include <hip/hip_runtime.h>

// Single-head causal attention, B=8 T=2048 C=1024 H=64.
// Kernel 1: fused QKV projection, fp32 -> f16 WMMA (f32 acc), outputs written
//           in WMMA-fragment-native swizzled layouts (q: A-layout prescaled,
//           k: B-layout over head dim, v: B-layout over key dim).
//           B-fragment LDS loads are batched ahead of the WMMA chain so the
//           compiler emits partial s_wait_dscnt values (pipelined LDS->WMMA).
// Kernel 2: flash attention, 1 wave per 16-query tile, 32-key blocks,
//           online softmax, P transposed via LDS, all GEMMs on v_wmma.
//           V fragments are loaded before the softmax VALU work; next block's
//           K/V fragments are prefetched (global_prefetch_b8).

typedef _Float16 f16;
typedef __attribute__((ext_vector_type(16))) _Float16 v16h;
typedef __attribute__((ext_vector_type(8)))  _Float16 v8h;
typedef __attribute__((ext_vector_type(8)))  float    v8f;
typedef __attribute__((ext_vector_type(4)))  float    v4f;

union V16 { v16h v; v8h h[2]; };

#define BB 8
#define TT 2048
#define CC 1024
#define HH 64

__device__ __forceinline__ v8f zero8() {
  v8f z = {0.f, 0.f, 0.f, 0.f, 0.f, 0.f, 0.f, 0.f};
  return z;
}

// ---------------------------------------------------------------------------
// Kernel 1: QKV projection. 128 blocks x 256 threads (8 waves).
// Each wave owns a 16-row tile of x; block stages W^T k-slices in LDS.
// ---------------------------------------------------------------------------
__global__ __launch_bounds__(256) void qkv_proj_wmma(
    const float* __restrict__ x, const float* __restrict__ Wq,
    const float* __restrict__ Wk, const float* __restrict__ Wv,
    f16* __restrict__ qA, f16* __restrict__ kB, f16* __restrict__ vB)
{
  __shared__ __align__(16) unsigned char smem[49152];  // union: W^T stage / out stage
  f16* wT = (f16*)smem;                                // [3][64 h][32 k] f16 = 12 KB

  const int tid  = threadIdx.x;
  const int lane = tid & 31;
  const int wave = tid >> 5;
  const int ln   = lane & 15;
  const int hiL  = lane >> 4;                 // 0: lanes 0-15, 1: lanes 16-31
  const int rowBase = blockIdx.x * 128;       // flat row in [0, B*T)
  const int waveRow = rowBase + wave * 16;

  v8f acc[12];
#pragma unroll
  for (int i = 0; i < 12; ++i) acc[i] = zero8();

  for (int k0 = 0; k0 < CC; k0 += 32) {
    __syncthreads();
    // Stage transposed W slice: wT[m][h][kk] = W[m][k0+kk][h]  (f32 -> f16)
    for (int i = tid; i < 3 * 2048; i += 256) {
      const int m  = i >> 11;
      const int j  = i & 2047;
      const int kk = j >> 6;
      const int h  = j & 63;
      const float* wp = (m == 0) ? Wq : ((m == 1) ? Wk : Wv);
      wT[m * 2048 + h * 32 + kk] = (f16)wp[(k0 + kk) * HH + h];
    }
    __syncthreads();

    // A fragment (16x32 f16) straight from x with b128 loads + cvt
    const float* xr = x + (size_t)(waveRow + ln) * CC + k0 + (hiL ? 8 : 0);
    v4f a0 = *(const v4f*)(xr + 0);
    v4f a1 = *(const v4f*)(xr + 4);
    v4f a2 = *(const v4f*)(xr + 16);
    v4f a3 = *(const v4f*)(xr + 20);
    V16 a;
#pragma unroll
    for (int i = 0; i < 4; ++i) {
      a.v[i]      = (f16)a0[i];
      a.v[4 + i]  = (f16)a1[i];
      a.v[8 + i]  = (f16)a2[i];
      a.v[12 + i] = (f16)a3[i];
    }

    // Batch all 12 B-fragment loads (24x ds_load_b128) before any WMMA so
    // dscnt waits are partial and LDS traffic overlaps matrix issue.
    V16 b[12];
#pragma unroll
    for (int m = 0; m < 3; ++m)
#pragma unroll
      for (int t = 0; t < 4; ++t) {
        const v8h* bp = (const v8h*)(wT + m * 2048 + (t * 16 + ln) * 32 + (hiL ? 8 : 0));
        b[m * 4 + t].h[0] = bp[0];
        b[m * 4 + t].h[1] = bp[2];
      }
#pragma unroll
    for (int i = 0; i < 12; ++i)
      acc[i] = __builtin_amdgcn_wmma_f32_16x16x32_f16(
          false, a.v, false, b[i].v, (short)0, acc[i], false, false);
  }

  // ---- epilogue: restage D tiles (C-layout) in LDS, emit fragment layouts ----
  __syncthreads();
  f16* qS = (f16*)smem;            // [8][16][64]
  f16* kS = qS + 8 * 1024;
  f16* vS = kS + 8 * 1024;         // total 48 KB
#pragma unroll
  for (int m = 0; m < 3; ++m) {
    f16* S = (m == 0) ? qS : ((m == 1) ? kS : vS);
#pragma unroll
    for (int t = 0; t < 4; ++t)
#pragma unroll
      for (int r = 0; r < 8; ++r) {
        float val = acc[m * 4 + t][r];
        if (m == 0) val *= 0.125f;                  // q pre-scaled by H^-0.5
        const int row = r + (hiL ? 8 : 0);
        S[wave * 1024 + row * 64 + t * 16 + ln] = (f16)val;
      }
  }
  __syncthreads();

  const int flatTile = waveRow >> 4;
  v8h* qD = (v8h*)qA;
  v8h* kD = (v8h*)kB;
  v8h* vD = (v8h*)vB;

  // q: A-layout fragments, 2 head-steps of K=32
#pragma unroll
  for (int s = 0; s < 2; ++s) {
    const f16* base = qS + wave * 1024 + ln * 64 + s * 32 + (hiL ? 8 : 0);
    v8h c0 = *(const v8h*)(base);
    v8h c1 = *(const v8h*)(base + 16);
    qD[((flatTile * 2 + s) * 2 + 0) * 32 + lane] = c0;
    qD[((flatTile * 2 + s) * 2 + 1) * 32 + lane] = c1;
  }
  // k: B-layout fragments for S = Q K^T (K-dim = head)
#pragma unroll
  for (int s = 0; s < 2; ++s) {
    const f16* base = kS + wave * 1024 + ln * 64 + s * 32 + (hiL ? 8 : 0);
    v8h c0 = *(const v8h*)(base);
    v8h c1 = *(const v8h*)(base + 16);
    kD[((flatTile * 2 + s) * 2 + 0) * 32 + lane] = c0;
    kD[((flatTile * 2 + s) * 2 + 1) * 32 + lane] = c1;
  }
  // v: B-layout fragments for O = P V (K-dim = 32 keys spanning 2 wave tiles)
#pragma unroll
  for (int j = 0; j < 2; ++j) {
    const int f    = wave * 2 + j;      // 16 fragments per block
    const int t32l = f >> 2;
    const int ht   = f & 3;
    const int h    = ht * 16 + ln;
    const int kb0  = hiL ? 16 : 0;
    V16 frag;
#pragma unroll
    for (int kk = 0; kk < 16; ++kk)
      frag.v[kk] = vS[(t32l * 32 + kb0 + kk) * 64 + h];
    const int fragIdx = ((rowBase >> 5) + t32l) * 4 + ht;
    vD[(fragIdx * 2 + 0) * 32 + lane] = frag.h[0];
    vD[(fragIdx * 2 + 1) * 32 + lane] = frag.h[1];
  }
}

// ---------------------------------------------------------------------------
// Kernel 2: flash attention. 128 blocks x 256 threads; 1 wave = 16 queries.
// ---------------------------------------------------------------------------
__global__ __launch_bounds__(256) void flash_attn_wmma(
    const f16* __restrict__ qA, const f16* __restrict__ kB,
    const f16* __restrict__ vB, float* __restrict__ out)
{
  __shared__ __align__(16) f16 pS[8 * 16 * 32];   // per-wave 16x32 P tile (8 KB)
  const int tid  = threadIdx.x;
  const int lane = tid & 31;
  const int wave = tid >> 5;
  const int ln   = lane & 15;
  const int hiL  = lane >> 4;

  const int flatTile = blockIdx.x * 8 + wave;     // [0, 1024)
  const int b     = flatTile >> 7;                // / (T/16)
  const int qBase = (flatTile & 127) * 16;

  const v8h* qP = (const v8h*)qA;
  const v8h* kP = (const v8h*)kB;
  const v8h* vP = (const v8h*)vB;

  V16 qa[2];
#pragma unroll
  for (int s = 0; s < 2; ++s) {
    qa[s].h[0] = qP[((flatTile * 2 + s) * 2 + 0) * 32 + lane];
    qa[s].h[1] = qP[((flatTile * 2 + s) * 2 + 1) * 32 + lane];
  }

  v8f o[4];
#pragma unroll
  for (int t = 0; t < 4; ++t) o[t] = zero8();
  float mr[8], lr[8];
#pragma unroll
  for (int r = 0; r < 8; ++r) { mr[r] = -3.0e38f; lr[r] = 0.0f; }

  f16* myP = pS + wave * 512;
  const int nb = (qBase + 15) / 32 + 1;           // causal block count

  for (int kb = 0; kb < nb; ++kb) {
    // Prefetch next key block's K/V fragments toward the WGP (L2-resident).
    if (kb + 1 < nb) {
      const int ktn = b * 128 + (kb + 1) * 2;
      __builtin_prefetch((const void*)&kP[(ktn * 4) * 32 + lane], 0, 0);
      __builtin_prefetch((const void*)&vP[(((b * 64 + kb + 1) * 4) * 2) * 32 + lane], 0, 0);
    }

    // K fragments + S GEMMs
    v8f s0 = zero8(), s1 = zero8();
#pragma unroll
    for (int hs = 0; hs < 2; ++hs) {
      const int kt0 = b * 128 + kb * 2;
      V16 k0f, k1f;
      k0f.h[0] = kP[(((kt0 + 0) * 2 + hs) * 2 + 0) * 32 + lane];
      k0f.h[1] = kP[(((kt0 + 0) * 2 + hs) * 2 + 1) * 32 + lane];
      k1f.h[0] = kP[(((kt0 + 1) * 2 + hs) * 2 + 0) * 32 + lane];
      k1f.h[1] = kP[(((kt0 + 1) * 2 + hs) * 2 + 1) * 32 + lane];
      s0 = __builtin_amdgcn_wmma_f32_16x16x32_f16(false, qa[hs].v, false, k0f.v,
                                                  (short)0, s0, false, false);
      s1 = __builtin_amdgcn_wmma_f32_16x16x32_f16(false, qa[hs].v, false, k1f.v,
                                                  (short)0, s1, false, false);
    }

    // V fragments issued now: loadcnt drains under the softmax VALU work below.
    V16 vf[4];
#pragma unroll
    for (int t = 0; t < 4; ++t) {
      const int fragIdx = (b * 64 + kb) * 4 + t;
      vf[t].h[0] = vP[(fragIdx * 2 + 0) * 32 + lane];
      vf[t].h[1] = vP[(fragIdx * 2 + 1) * 32 + lane];
    }

    if (kb * 32 + 31 > qBase) {                   // diagonal block: causal mask
#pragma unroll
      for (int r = 0; r < 8; ++r) {
        const int qrow = qBase + r + (hiL ? 8 : 0);
        const int key  = kb * 32 + ln;
        s0[r] = (key      > qrow) ? -3.0e38f : s0[r];
        s1[r] = (key + 16 > qrow) ? -3.0e38f : s1[r];
      }
    }

    float al[8];
#pragma unroll
    for (int r = 0; r < 8; ++r) {
      float vmax = fmaxf(s0[r], s1[r]);
#pragma unroll
      for (int d = 1; d < 16; d <<= 1)
        vmax = fmaxf(vmax, __shfl_xor(vmax, d, 16));
      const float mnew = fmaxf(mr[r], vmax);
      al[r] = __expf(mr[r] - mnew);
      const float p0 = __expf(s0[r] - mnew);
      const float p1 = __expf(s1[r] - mnew);
      float rs = p0 + p1;
#pragma unroll
      for (int d = 1; d < 16; d <<= 1)
        rs += __shfl_xor(rs, d, 16);
      lr[r] = lr[r] * al[r] + rs;
      mr[r] = mnew;
      s0[r] = p0;
      s1[r] = p1;
    }
#pragma unroll
    for (int t = 0; t < 4; ++t)
#pragma unroll
      for (int r = 0; r < 8; ++r) o[t][r] *= al[r];

    // Transpose P (C-layout f32 -> A-layout f16) through private LDS tile
#pragma unroll
    for (int r = 0; r < 8; ++r) {
      const int row = r + (hiL ? 8 : 0);
      myP[row * 32 + ln]      = (f16)s0[r];
      myP[row * 32 + 16 + ln] = (f16)s1[r];
    }
    asm volatile("s_wait_dscnt 0" ::: "memory");  // per-wave DS RAW fence
    V16 pa;
    const f16* pb = myP + ln * 32 + (hiL ? 16 : 0);
    pa.h[0] = *(const v8h*)(pb);
    pa.h[1] = *(const v8h*)(pb + 8);

#pragma unroll
    for (int t = 0; t < 4; ++t)
      o[t] = __builtin_amdgcn_wmma_f32_16x16x32_f16(false, pa.v, false, vf[t].v,
                                                    (short)0, o[t], false, false);
  }

#pragma unroll
  for (int r = 0; r < 8; ++r) lr[r] = 1.0f / lr[r];
#pragma unroll
  for (int t = 0; t < 4; ++t)
#pragma unroll
    for (int r = 0; r < 8; ++r) {
      const int row = qBase + r + (hiL ? 8 : 0);
      out[((size_t)b * TT + row) * HH + t * 16 + ln] = o[t][r] * lr[r];
    }
}

// ---------------------------------------------------------------------------
extern "C" void kernel_launch(void* const* d_in, const int* in_sizes, int n_in,
                              void* d_out, int out_size, void* d_ws, size_t ws_size,
                              hipStream_t stream) {
  const float* x  = (const float*)d_in[0];
  const float* Wq = (const float*)d_in[1];
  const float* Wk = (const float*)d_in[2];
  const float* Wv = (const float*)d_in[3];

  // workspace: q (A-layout), k (B-layout), v (B-layout), each B*T*H f16 = 2 MB
  f16* qA = (f16*)d_ws;
  f16* kBp = qA  + (size_t)BB * TT * HH;
  f16* vBp = kBp + (size_t)BB * TT * HH;

  qkv_proj_wmma<<<128, 256, 0, stream>>>(x, Wq, Wk, Wv, qA, kBp, vBp);
  flash_attn_wmma<<<128, 256, 0, stream>>>(qA, kBp, vBp, (float*)d_out);
}
